// SpatialEvoConv_25890062860993
// MI455X (gfx1250) — matrix operand order repaired and checked by reading, not compiled, not run on hardware
//
#include <hip/hip_runtime.h>

// ---- problem constants (match reference) ----
#define NN      20000
#define OUT64   64
#define DDIM32  32
#define EREL    150000
#define EFLOW   40000
#define ECAT    120000          // 3*EFLOW
#define NB      15
#define ROW64   64
#define N64     (NN * 64)

typedef __bf16 bhalf;
typedef bhalf v16bf __attribute__((ext_vector_type(16)));
typedef float v8f   __attribute__((ext_vector_type(8)));

__device__ inline v8f zero8() { v8f z = {0.f,0.f,0.f,0.f,0.f,0.f,0.f,0.f}; return z; }

__device__ inline v8f wmma_bf(v16bf a, v16bf b, v8f c) {
    return __builtin_amdgcn_wmma_f32_16x16x32_bf16(false, a, false, b, (short)0, c, false, false);
}

__device__ inline void wave_lds_fence() {
    asm volatile("s_wait_dscnt 0x0" ::: "memory");
}

__device__ inline float atomicAddF(float* p, float v) {
    return __hip_atomic_fetch_add(p, v, __ATOMIC_RELAXED, __HIP_MEMORY_SCOPE_AGENT);
}

// A fragment (16x32 bf16) from an f32 row (this lane supplies row m = lane&15).
// lane<16: K = kb+{0..7, 16..23}; lane>=16: K = kb+{8..15, 24..31}
__device__ inline v16bf afrag_f32(const float* __restrict__ row, int kb, int lane) {
    const float* p = row + kb + ((lane & 16) ? 8 : 0);
    v16bf a;
#pragma unroll
    for (int q = 0; q < 8; ++q) a[q] = (bhalf)p[q];
#pragma unroll
    for (int q = 0; q < 8; ++q) a[8 + q] = (bhalf)p[16 + q];
    return a;
}

// A fragment from sum of three f32 rows
__device__ inline v16bf afrag_sum3(const float* __restrict__ r0, const float* __restrict__ r1,
                                   const float* __restrict__ r2, int kb, int lane) {
    int o = kb + ((lane & 16) ? 8 : 0);
    v16bf a;
#pragma unroll
    for (int q = 0; q < 8; ++q) a[q] = (bhalf)(r0[o + q] + r1[o + q] + r2[o + q]);
#pragma unroll
    for (int q = 0; q < 8; ++q) a[8 + q] = (bhalf)(r0[o + 16 + q] + r1[o + 16 + q] + r2[o + 16 + q]);
    return a;
}

// A fragment from a bf16 LDS row
__device__ inline v16bf afrag_lds(const bhalf* __restrict__ row, int kb, int lane) {
    const bhalf* p = row + kb + ((lane & 16) ? 8 : 0);
    v16bf a;
#pragma unroll
    for (int q = 0; q < 8; ++q) a[q] = p[q];
#pragma unroll
    for (int q = 0; q < 8; ++q) a[8 + q] = p[16 + q];
    return a;
}

// B fragment (32x16 bf16) from bf16 weights stored [n][k] row-major with leading dim ld.
// lane supplies column n = 16*ntile + (lane&15); lane<16: K = kb+{0..7, 16..23}; lane>=16: K = kb+{8..15, 24..31}
__device__ inline v16bf bfragld(const bhalf* __restrict__ W, int ld, int kb, int ntile, int lane) {
    const bhalf* p = W + (16 * ntile + (lane & 15)) * ld + kb + ((lane & 16) ? 8 : 0);
    v16bf b;
#pragma unroll
    for (int q = 0; q < 8; ++q) b[q] = p[q];
#pragma unroll
    for (int q = 0; q < 8; ++q) b[8 + q] = p[16 + q];
    return b;
}

__device__ inline int bucketize(float d, const float* bl) {
    int c = 0;
#pragma unroll
    for (int j = 0; j < NB; ++j) c += (bl[j] < d) ? 1 : 0;
    return c;
}

// ---------------- weight fp32 -> bf16 conversion ----------------
__global__ __launch_bounds__(256) void k_prep(
    const float* __restrict__ fc2, const float* __restrict__ fcd,
    const float* __restrict__ w1, const float* __restrict__ w2,
    const float* __restrict__ sp, const float* __restrict__ agg, const float* __restrict__ G,
    bhalf* fc2b, bhalf* fcdb, bhalf* w1b, bhalf* w2b, bhalf* spb, bhalf* aggb, bhalf* Gb)
{
    int id = blockIdx.x * 256 + threadIdx.x;
    const int s0 = 4*64*64, s1 = 4*64*32, s2 = 4*64*128, s3 = 4*64*128, s4 = 64*64, s5 = 64*128, s6 = 64*32;
    if (id < s0) { fc2b[id] = (bhalf)fc2[id]; return; }  id -= s0;
    if (id < s1) { fcdb[id] = (bhalf)fcd[id]; return; }  id -= s1;
    if (id < s2) { w1b[id]  = (bhalf)w1[id];  return; }  id -= s2;
    if (id < s3) { w2b[id]  = (bhalf)w2[id];  return; }  id -= s3;
    if (id < s4) { spb[id]  = (bhalf)sp[id];  return; }  id -= s4;
    if (id < s5) { aggb[id] = (bhalf)agg[id]; return; }  id -= s5;
    if (id < s6) { Gb[id]   = (bhalf)G[id];   return; }
}

// ---------------- h2[r] = feat @ fc2[r]^T ----------------
__global__ __launch_bounds__(256) void k_h2(const float* __restrict__ feat,
                                            const bhalf* __restrict__ fc2b,
                                            float* __restrict__ h2)
{
    const int lane = threadIdx.x & 31, w = threadIdx.x >> 5;
    const int r = blockIdx.y;
    const int tile = blockIdx.x * 8 + w;
    if (tile >= NN / 16) return;
    const int m = lane & 15;
    const float* row = feat + (tile * 16 + m) * 64;
    const bhalf* W = fc2b + r * 64 * 64;
    v8f acc[4] = {zero8(), zero8(), zero8(), zero8()};
#pragma unroll
    for (int ks = 0; ks < 2; ++ks) {
        v16bf a = afrag_f32(row, ks * 32, lane);
#pragma unroll
        for (int jj = 0; jj < 4; ++jj) acc[jj] = wmma_bf(a, bfragld(W, 64, ks * 32, jj, lane), acc[jj]);
    }
    float* o = h2 + r * N64;
    const int rb = (lane & 16) ? 8 : 0;
#pragma unroll
    for (int jj = 0; jj < 4; ++jj)
#pragma unroll
        for (int v = 0; v < 8; ++v)
            o[(tile * 16 + v + rb) * 64 + (lane & 15) + 16 * jj] = acc[jj][v];
}

// ---------------- degrees, phase 1 (per etype) ----------------
__global__ __launch_bounds__(256) void k_deg1(const int* __restrict__ rsrc, const int* __restrict__ rdst,
                                              float* __restrict__ din1, float* __restrict__ dout1)
{
    int et = blockIdx.y;
    int e = blockIdx.x * 256 + threadIdx.x;
    if (e >= EREL) return;
    atomicAddF(din1  + et * NN + rdst[et * EREL + e], 1.f);
    atomicAddF(dout1 + et * NN + rsrc[et * EREL + e], 1.f);
}

// ---------------- degrees, phase 2 (per flow set) ----------------
__global__ __launch_bounds__(256) void k_fdeg(const int* __restrict__ fsrc, const int* __restrict__ fdst,
                                              float* __restrict__ fdin, float* __restrict__ fdout)
{
    int t = blockIdx.y;
    int e = blockIdx.x * 256 + threadIdx.x;
    if (e >= EFLOW) return;
    atomicAddF(fdin  + t * NN + fdst[t * EFLOW + e], 1.f);
    atomicAddF(fdout + t * NN + fsrc[t * EFLOW + e], 1.f);
}

// ---------------- phase 1 edge kernel ----------------
__global__ __launch_bounds__(256) void k_twohop(
    const float* __restrict__ h2, const float* __restrict__ loc,
    const float* __restrict__ bnd, const float* __restrict__ embed,
    const bhalf* __restrict__ fcdb, const bhalf* __restrict__ w1b, const bhalf* __restrict__ w2b,
    const float* __restrict__ vaw,
    const int* __restrict__ rsrc, const int* __restrict__ rdst, const int* __restrict__ rmid,
    const float* __restrict__ din1, const float* __restrict__ dout1,
    float* __restrict__ hlist)
{
    __shared__ bhalf stage[8][16 * 64];
    __shared__ float betaL[8][16], normL[8][16];
    __shared__ int eSrc[8][16], eDst[8][16], eMid[8][16];

    const int lane = threadIdx.x & 31, w = threadIdx.x >> 5;
    const int et = blockIdx.y, t = et >> 2, r = et & 3;
    const int tile = blockIdx.x * 8 + w;
    if (tile >= EREL / 16) return;
    const int m = lane & 15;
    const int e = tile * 16 + m;
    const int base = et * EREL;
    const int src = rsrc[base + e], dst = rdst[base + e], mid = rmid[base + e];
    const float* h2r = h2 + r * N64;

    float bl[NB];
#pragma unroll
    for (int j = 0; j < NB; ++j) bl[j] = bnd[j];

    if (lane < 16) {
        eSrc[w][m] = src; eDst[w][m] = dst; eMid[w][m] = mid;
        normL[w][m] = rsqrtf(fmaxf(din1[et * NN + dst], 1.f)) *
                      rsqrtf(fmaxf(dout1[et * NN + src], 1.f));
    }

    float dx = loc[2 * dst] - loc[2 * src], dy = loc[2 * dst + 1] - loc[2 * src + 1];
    const float* erow = embed + bucketize(sqrtf(dx * dx + dy * dy), bl) * DDIM32;

    // de = embed @ fcd^T  (K=32)
    v8f accde[4] = {zero8(), zero8(), zero8(), zero8()};
    {
        v16bf a = afrag_f32(erow, 0, lane);
#pragma unroll
        for (int jj = 0; jj < 4; ++jj)
            accde[jj] = wmma_bf(a, bfragld(fcdb + r * 64 * 32, 32, 0, jj, lane), accde[jj]);
    }
    const int rb = (lane & 16) ? 8 : 0;
#pragma unroll
    for (int jj = 0; jj < 4; ++jj)
#pragma unroll
        for (int v = 0; v < 8; ++v)
            stage[w][(v + rb) * 64 + (lane & 15) + 16 * jj] = (bhalf)accde[jj][v];
    wave_lds_fence();

    // acc = concat(h2[dst], de) @ w1^T + concat(h2[mid], h2[src]) @ w2^T
    v8f acc[4] = {zero8(), zero8(), zero8(), zero8()};
    const bhalf* W1 = w1b + r * 64 * 128;
    const bhalf* W2 = w2b + r * 64 * 128;
    const float* rowd = h2r + dst * 64;
    const float* rowm = h2r + mid * 64;
    const float* rows = h2r + src * 64;
    const bhalf* srow = &stage[w][(lane & 15) * 64];
#pragma unroll
    for (int ks = 0; ks < 2; ++ks) {
        int kb = ks * 32;
        v16bf a = afrag_f32(rowd, kb, lane);
#pragma unroll
        for (int jj = 0; jj < 4; ++jj) acc[jj] = wmma_bf(a, bfragld(W1, 128, kb, jj, lane), acc[jj]);
    }
#pragma unroll
    for (int ks = 0; ks < 2; ++ks) {
        int kb = ks * 32;
        v16bf a = afrag_lds(srow, kb, lane);
#pragma unroll
        for (int jj = 0; jj < 4; ++jj) acc[jj] = wmma_bf(a, bfragld(W1, 128, 64 + kb, jj, lane), acc[jj]);
    }
#pragma unroll
    for (int ks = 0; ks < 2; ++ks) {
        int kb = ks * 32;
        v16bf a = afrag_f32(rowm, kb, lane);
#pragma unroll
        for (int jj = 0; jj < 4; ++jj) acc[jj] = wmma_bf(a, bfragld(W2, 128, kb, jj, lane), acc[jj]);
    }
#pragma unroll
    for (int ks = 0; ks < 2; ++ks) {
        int kb = ks * 32;
        v16bf a = afrag_f32(rows, kb, lane);
#pragma unroll
        for (int jj = 0; jj < 4; ++jj) acc[jj] = wmma_bf(a, bfragld(W2, 128, 64 + kb, jj, lane), acc[jj]);
    }

    // beta[m] = sigmoid(sum_n acc[m][n] * va[n])   (cross-lane reduce in 16-lane halves)
    const float* va = vaw + r * 64;
    float p[8];
#pragma unroll
    for (int v = 0; v < 8; ++v) {
        float s = 0.f;
#pragma unroll
        for (int jj = 0; jj < 4; ++jj) s += acc[jj][v] * va[16 * jj + (lane & 15)];
        p[v] = s;
    }
#pragma unroll
    for (int mask = 1; mask <= 8; mask <<= 1)
#pragma unroll
        for (int v = 0; v < 8; ++v) p[v] += __shfl_xor(p[v], mask, 32);
    if ((lane & 15) == 0) {
#pragma unroll
        for (int v = 0; v < 8; ++v) betaL[w][v + rb] = 1.f / (1.f + expf(-p[v]));
    }
    wave_lds_fence();

    // epilogue: he = (beta*h2[src] + h2[mid]) * norm, scatter-add into hlist[t]
    float* hl = hlist + t * N64;
#pragma unroll
    for (int e2 = 0; e2 < 16; ++e2) {
        float be = betaL[w][e2], no = normL[w][e2];
        const float* rs = h2r + eSrc[w][e2] * 64;
        const float* rm = h2r + eMid[w][e2] * 64;
        float* od = hl + eDst[w][e2] * 64;
        int n = lane;
        atomicAddF(od + n, (be * rs[n] + rm[n]) * no);
        n = lane + 32;
        atomicAddF(od + n, (be * rs[n] + rm[n]) * no);
    }
}

// ---------------- f[i] = (hlist[i-1]+hlist[i]+hlist[i+1]) @ sp_fc^T ----------------
__global__ __launch_bounds__(256) void k_fsum(const float* __restrict__ hlist,
                                              const bhalf* __restrict__ spb,
                                              float* __restrict__ f)
{
    const int lane = threadIdx.x & 31, w = threadIdx.x >> 5;
    const int i = blockIdx.y, i1 = (i + 3) & 3, i2 = (i + 1) & 3;
    const int tile = blockIdx.x * 8 + w;
    if (tile >= NN / 16) return;
    const int m = lane & 15;
    const int node = tile * 16 + m;
    const float* r0 = hlist + i1 * N64 + node * 64;
    const float* r1 = hlist + i  * N64 + node * 64;
    const float* r2 = hlist + i2 * N64 + node * 64;
    v8f acc[4] = {zero8(), zero8(), zero8(), zero8()};
#pragma unroll
    for (int ks = 0; ks < 2; ++ks) {
        v16bf a = afrag_sum3(r0, r1, r2, ks * 32, lane);
#pragma unroll
        for (int jj = 0; jj < 4; ++jj) acc[jj] = wmma_bf(a, bfragld(spb, 64, ks * 32, jj, lane), acc[jj]);
    }
    float* o = f + i * N64;
    const int rb = (lane & 16) ? 8 : 0;
#pragma unroll
    for (int jj = 0; jj < 4; ++jj)
#pragma unroll
        for (int v = 0; v < 8; ++v)
            o[(tile * 16 + v + rb) * 64 + (lane & 15) + 16 * jj] = acc[jj][v];
}

// ---------------- phase 2 edge kernel ----------------
__global__ __launch_bounds__(256) void k_seprop(
    const float* __restrict__ f, const float* __restrict__ loc,
    const float* __restrict__ bnd, const float* __restrict__ embed,
    const bhalf* __restrict__ Gb, const bhalf* __restrict__ aggbf, const float* __restrict__ aggB,
    const int* __restrict__ fsrc, const int* __restrict__ fdst, const int* __restrict__ inter,
    const float* __restrict__ fdin, const float* __restrict__ fdout,
    float* __restrict__ out)
{
    __shared__ bhalf stage[8][16 * 128];
    __shared__ int eSrc[8][16], eDst[8][16], eInt[8][16 * 5];
    __shared__ float normL[8][16];

    const int lane = threadIdx.x & 31, w = threadIdx.x >> 5;
    const int i = blockIdx.y, i1 = (i + 3) & 3, i2 = (i + 1) & 3;
    const int tile = blockIdx.x * 8 + w;
    if (tile >= ECAT / 16) return;
    const int m = lane & 15;
    const int e = tile * 16 + m;
    const int seg = e / EFLOW;
    const int local = e - seg * EFLOW;
    const int ts = (seg == 0) ? i : ((seg == 1) ? i1 : i2);
    const int src = fsrc[ts * EFLOW + local];
    const int dst = fdst[ts * EFLOW + local];
    int itj[5];
#pragma unroll
    for (int j = 0; j < 5; ++j) itj[j] = inter[((size_t)i * ECAT + e) * 5 + j];

    float bl[NB];
#pragma unroll
    for (int j = 0; j < NB; ++j) bl[j] = bnd[j];

    if (lane < 16) {
        eSrc[w][m] = src; eDst[w][m] = dst;
#pragma unroll
        for (int j = 0; j < 5; ++j) eInt[w][m * 5 + j] = itj[j];
        float di = fdin[i1 * NN + dst] + fdin[i * NN + dst] + fdin[i2 * NN + dst];
        float dn = fdout[i1 * NN + src] + fdout[i * NN + src] + fdout[i2 * NN + src];
        normL[w][m] = rsqrtf(fmaxf(di, 1.f)) * rsqrtf(fmaxf(dn, 1.f));
    }
    wave_lds_fence();

    const float* fi = f + i * N64;
    const float sx = loc[2 * src], sy = loc[2 * src + 1];
    const int rb = (lane & 16) ? 8 : 0;

    // e1 = embed(dist(src,dst)) @ G^T ; stage left half = e1 * f[src]
    {
        float dx = loc[2 * dst] - sx, dy = loc[2 * dst + 1] - sy;
        const float* erow = embed + bucketize(sqrtf(dx * dx + dy * dy), bl) * DDIM32;
        v16bf a = afrag_f32(erow, 0, lane);
        v8f a1[4] = {zero8(), zero8(), zero8(), zero8()};
#pragma unroll
        for (int jj = 0; jj < 4; ++jj) a1[jj] = wmma_bf(a, bfragld(Gb, 32, 0, jj, lane), a1[jj]);
#pragma unroll
        for (int jj = 0; jj < 4; ++jj)
#pragma unroll
            for (int v = 0; v < 8; ++v) {
                int m2 = v + rb, n = (lane & 15) + 16 * jj;
                float fv = fi[eSrc[w][m2] * 64 + n];
                stage[w][m2 * 128 + n] = (bhalf)(a1[jj][v] * fv);
            }
    }

    // dot_h = mean_j (embed(dist(src,inter_j)) @ G^T) * f[inter_j] ; stage right half
    {
        v8f dacc[4] = {zero8(), zero8(), zero8(), zero8()};
#pragma unroll
        for (int j = 0; j < 5; ++j) {
            int it = itj[j];
            float dx = sx - loc[2 * it], dy = sy - loc[2 * it + 1];
            const float* erow = embed + bucketize(sqrtf(dx * dx + dy * dy), bl) * DDIM32;
            v16bf a = afrag_f32(erow, 0, lane);
            v8f ej[4] = {zero8(), zero8(), zero8(), zero8()};
#pragma unroll
            for (int jj = 0; jj < 4; ++jj) ej[jj] = wmma_bf(a, bfragld(Gb, 32, 0, jj, lane), ej[jj]);
#pragma unroll
            for (int jj = 0; jj < 4; ++jj)
#pragma unroll
                for (int v = 0; v < 8; ++v) {
                    int m2 = v + rb, n = (lane & 15) + 16 * jj;
                    float fv = fi[eInt[w][m2 * 5 + j] * 64 + n];
                    dacc[jj][v] += ej[jj][v] * fv;
                }
        }
#pragma unroll
        for (int jj = 0; jj < 4; ++jj)
#pragma unroll
            for (int v = 0; v < 8; ++v) {
                int m2 = v + rb, n = (lane & 15) + 16 * jj;
                stage[w][m2 * 128 + 64 + n] = (bhalf)(dacc[jj][v] * 0.2f);
            }
    }
    wave_lds_fence();

    // x = stage(16x128) @ agg^T + agg_b ; out += norm * x (scatter)
    v8f ax[4] = {zero8(), zero8(), zero8(), zero8()};
    const bhalf* srow = &stage[w][(lane & 15) * 128];
#pragma unroll
    for (int ks = 0; ks < 4; ++ks) {
        int kb = ks * 32;
        v16bf a = afrag_lds(srow, kb, lane);
#pragma unroll
        for (int jj = 0; jj < 4; ++jj) ax[jj] = wmma_bf(a, bfragld(aggbf, 128, kb, jj, lane), ax[jj]);
    }
#pragma unroll
    for (int jj = 0; jj < 4; ++jj)
#pragma unroll
        for (int v = 0; v < 8; ++v) {
            int m2 = v + rb, n = (lane & 15) + 16 * jj;
            float val = (ax[jj][v] + aggB[n]) * normL[w][m2];
            atomicAddF(out + ((size_t)i * NN + eDst[w][m2]) * 64 + n, val);
        }
}

// ---------------- host launcher ----------------
extern "C" void kernel_launch(void* const* d_in, const int* in_sizes, int n_in,
                              void* d_out, int out_size, void* d_ws, size_t ws_size,
                              hipStream_t stream) {
    const float* feat  = (const float*)d_in[0];
    const float* loc   = (const float*)d_in[1];
    const float* bnd   = (const float*)d_in[2];
    const float* embed = (const float*)d_in[3];
    const float* fc2   = (const float*)d_in[4];
    const float* fcd   = (const float*)d_in[5];
    const float* w1    = (const float*)d_in[6];
    const float* w2    = (const float*)d_in[7];
    const float* vaw   = (const float*)d_in[8];
    const float* spfc  = (const float*)d_in[9];
    const float* aggw  = (const float*)d_in[10];
    const float* aggb  = (const float*)d_in[11];
    const float* Gw    = (const float*)d_in[12];
    const int* rsrc    = (const int*)d_in[13];
    const int* rdst    = (const int*)d_in[14];
    const int* rmid    = (const int*)d_in[15];
    const int* fsrc    = (const int*)d_in[16];
    const int* fdst    = (const int*)d_in[17];
    const int* inter   = (const int*)d_in[18];
    float* outp = (float*)d_out;

    char* ws = (char*)d_ws;
    size_t off = 0;
    auto take = [&](size_t bytes) { char* p = ws + off; off += bytes; return p; };
    float* h2    = (float*)take((size_t)4 * N64 * 4);
    float* hlist = (float*)take((size_t)4 * N64 * 4);
    float* fbuf  = (float*)take((size_t)4 * N64 * 4);
    float* din1  = (float*)take((size_t)16 * NN * 4);
    float* dout1 = (float*)take((size_t)16 * NN * 4);
    float* fdin  = (float*)take((size_t)4 * NN * 4);
    float* fdout = (float*)take((size_t)4 * NN * 4);
    bhalf* fc2b  = (bhalf*)take(4 * 64 * 64 * 2);
    bhalf* fcdb  = (bhalf*)take(4 * 64 * 32 * 2);
    bhalf* w1b   = (bhalf*)take(4 * 64 * 128 * 2);
    bhalf* w2b   = (bhalf*)take(4 * 64 * 128 * 2);
    bhalf* spb   = (bhalf*)take(64 * 64 * 2);
    bhalf* aggbfp= (bhalf*)take(64 * 128 * 2);
    bhalf* Gb    = (bhalf*)take(64 * 32 * 2);

    hipMemsetAsync(hlist, 0, (size_t)4 * N64 * 4, stream);
    hipMemsetAsync(din1, 0, (size_t)16 * NN * 4, stream);
    hipMemsetAsync(dout1, 0, (size_t)16 * NN * 4, stream);
    hipMemsetAsync(fdin, 0, (size_t)4 * NN * 4, stream);
    hipMemsetAsync(fdout, 0, (size_t)4 * NN * 4, stream);
    hipMemsetAsync(outp, 0, (size_t)out_size * 4, stream);

    k_prep<<<408, 256, 0, stream>>>(fc2, fcd, w1, w2, spfc, aggw, Gw,
                                    fc2b, fcdb, w1b, w2b, spb, aggbfp, Gb);
    k_h2<<<dim3(157, 4), 256, 0, stream>>>(feat, fc2b, h2);
    k_deg1<<<dim3((EREL + 255) / 256, 16), 256, 0, stream>>>(rsrc, rdst, din1, dout1);
    k_fdeg<<<dim3((EFLOW + 255) / 256, 4), 256, 0, stream>>>(fsrc, fdst, fdin, fdout);
    k_twohop<<<dim3(1172, 16), 256, 0, stream>>>(h2, loc, bnd, embed, fcdb, w1b, w2b, vaw,
                                                 rsrc, rdst, rmid, din1, dout1, hlist);
    k_fsum<<<dim3(157, 4), 256, 0, stream>>>(hlist, spb, fbuf);
    k_seprop<<<dim3(938, 4), 256, 0, stream>>>(fbuf, loc, bnd, embed, Gb, aggbfp, aggb,
                                               fsrc, fdst, inter, fdin, fdout, outp);
    (void)n_in; (void)in_sizes; (void)ws_size;
}
